// FDConv_70420283785807
// MI455X (gfx1250) — matrix operand order repaired and checked by reading, not compiled, not run on hardware
//
#include <hip/hip_runtime.h>

#define B_    16
#define CIN_  128
#define COUT_ 128
#define H_    128
#define W_    128
#define KNUM_ 64

typedef __attribute__((ext_vector_type(16))) __bf16        v16bf;
typedef __attribute__((ext_vector_type(8)))  float         v8f;
typedef __attribute__((ext_vector_type(4)))  int           v4i_t;

union Frag { uint4 q[2]; v16bf b; };

#if __has_builtin(__builtin_amdgcn_global_load_async_to_lds_b128) && \
    __has_builtin(__builtin_amdgcn_s_wait_asynccnt)
#define USE_ASYNC_LDS 1
#endif

__device__ __forceinline__ unsigned short f2bf(float f) {
    unsigned u = __float_as_uint(f);
    u += 0x7FFFu + ((u >> 16) & 1u);     // round-to-nearest-even
    return (unsigned short)(u >> 16);
}

// ---------------- Stage 1: global average pool  pooled[b][c] ----------------
__global__ void fdconv_pool(const float* __restrict__ x, float* __restrict__ pooled) {
    __shared__ float red[256];
    int bc = blockIdx.x;                       // b*CIN + c
    const float* p = x + (size_t)bc * (H_ * W_);
    float s = 0.f;
    for (int i = threadIdx.x; i < H_ * W_; i += 256) s += p[i];
    red[threadIdx.x] = s;
    __syncthreads();
    for (int off = 128; off > 0; off >>= 1) {
        if (threadIdx.x < off) red[threadIdx.x] += red[threadIdx.x + off];
        __syncthreads();
    }
    if (threadIdx.x == 0) pooled[bc] = red[0] * (1.0f / (H_ * W_));
}

// ---------------- Stage 2: sigmoid(pooled @ ksm^T) -> softmax over K --------
__global__ void fdconv_att(const float* __restrict__ pooled,
                           const float* __restrict__ ksm,
                           float* __restrict__ att) {
    __shared__ float sm[KNUM_];
    int b = blockIdx.x, k = threadIdx.x;       // 64 threads
    const float* pb = pooled + b * CIN_;
    const float* wk = ksm + k * CIN_;
    float dot = 0.f;
    for (int c = 0; c < CIN_; ++c) dot += pb[c] * wk[c];
    float a = 1.f / (1.f + __expf(-dot));
    sm[k] = a;
    __syncthreads();
    float mx = -1e30f;
    for (int i = 0; i < KNUM_; ++i) mx = fmaxf(mx, sm[i]);
    float sum = 0.f;
    for (int i = 0; i < KNUM_; ++i) sum += __expf(sm[i] - mx);
    att[b * KNUM_ + k] = __expf(a - mx) / sum;
}

// ------- Stage 3: aggregate freq weights, 3x3 ortho IFFT (real), clip -------
// Output: bf16 kernels pre-packed in the WMMA A-fragment LDS layout:
//   per (b, co-tile of 16): ushort[18432], addr(k,m) = (k>>3)*128 + m*8 + (k&7),
//   with k = tap*128 + ci (tap-major K so 32-wide chunks never cross a tap).
__global__ void fdconv_agg_ifft(const float* __restrict__ att,
                                const float* __restrict__ dftw,
                                unsigned short* __restrict__ kerb) {
    __shared__ float att_s[KNUM_];
    int g   = blockIdx.x * 256 + threadIdx.x;  // b fixed within block (16384 % 256 == 0)
    int b   = g >> 14;
    int rem = g & 16383;
    int co  = rem >> 7;
    int ci  = rem & 127;
    if (threadIdx.x < KNUM_) att_s[threadIdx.x] = att[b * KNUM_ + threadIdx.x];
    __syncthreads();

    float re[9] = {0,0,0,0,0,0,0,0,0};
    float im[9] = {0,0,0,0,0,0,0,0,0};
    for (int k = 0; k < KNUM_; ++k) {
        float a = att_s[k];
        const float* w = dftw + (size_t)((k * COUT_ + co) * CIN_ + ci) * 18;
#pragma unroll
        for (int j = 0; j < 9; ++j) {
            re[j] += a * w[2 * j];
            im[j] += a * w[2 * j + 1];
        }
    }
    const float CT[3] = {1.f, -0.5f, -0.5f};
    const float ST[3] = {0.f, 0.8660254037844386f, -0.8660254037844386f};
    unsigned short* kout = kerb + (size_t)(b * 8 + (co >> 4)) * 18432;
    int mm = co & 15;
#pragma unroll
    for (int i = 0; i < 3; ++i)
#pragma unroll
        for (int j = 0; j < 3; ++j) {
            float s = 0.f;
#pragma unroll
            for (int u = 0; u < 3; ++u)
#pragma unroll
                for (int v = 0; v < 3; ++v) {
                    int ph = (i * u + j * v) % 3;   // exp(+2*pi*i*ph/3)
                    s += re[u * 3 + v] * CT[ph] - im[u * 3 + v] * ST[ph];
                }
            s *= (1.f / 3.f);                        // norm='ortho'
            s = fminf(10.f, fmaxf(-10.f, s));
            int t = i * 3 + j;
            int k = t * 128 + ci;
            kout[(k >> 3) * 128 + mm * 8 + (k & 7)] = f2bf(s);
        }
}

// ---- Stage 4: per-sample 3x3 conv as GEMM with v_wmma_f32_16x16x32_bf16 ----
// 128 thr (4 wave32) per block; tile = 16 co x 256 px (2 rows x 128 cols).
// Each wave owns 4 N-subtiles of 16 px sharing one A fragment per tap.
// A tile arrives pre-packed bf16 -> async global->LDS B128 copy (ASYNCcnt)
// when available; all fragment loads are ds_load_b128.
__global__ __launch_bounds__(128) void fdconv_conv_wmma(
    const float* __restrict__ x, const unsigned short* __restrict__ kerb,
    const float* __restrict__ bias, const float* __restrict__ fbm,
    float* __restrict__ out) {
    __shared__ unsigned int   AldsU[9216];       // 36 KB, [k/8][m=16][(k%8)/2] pairs
    __shared__ unsigned short Xs[4 * 130 * 32];  // 32.5 KB, [row 4][col 130][ci 32]

    int b     = blockIdx.z;
    int co0   = blockIdx.y * 16;
    int hrow0 = blockIdx.x * 2;

    int tid  = threadIdx.x;
    int lane = tid & 31;
    int wv   = tid >> 5;
    int h    = lane >> 4;                        // half: K sub-range select
    int m    = lane & 15;

    // Stage A tile: contiguous copy of pre-packed bf16 (2304 x 16B moves).
    const unsigned short* asrc = kerb + (size_t)(b * 8 + blockIdx.y) * 18432;
#if defined(USE_ASYNC_LDS)
    {
        v4i_t* g = (v4i_t*)asrc + tid;           // 16B elements, lane-strided
        v4i_t* l = (v4i_t*)AldsU + tid;
#pragma unroll
        for (int i = 0; i < 18; ++i)             // 18 * 128 lanes * 16B = 36 KB
            __builtin_amdgcn_global_load_async_to_lds_b128(
                g + i * 128, l + i * 128, 0, 0);
    }
#else
    {
        const uint4* g4 = (const uint4*)asrc;
        uint4* l4 = (uint4*)AldsU;
        for (int i = tid; i < 2304; i += 128) l4[i] = g4[i];
    }
#endif

    // Per-lane pixel coordinates for the 4 N-subtiles of this wave.
    int prow[4], pcol[4];
#pragma unroll
    for (int n = 0; n < 4; ++n) {
        int p = wv * 64 + n * 16 + m;            // 0..255 within tile
        prow[n] = p >> 7;
        pcol[n] = p & 127;
    }

#if defined(USE_ASYNC_LDS)
    __builtin_amdgcn_s_wait_asynccnt(0);         // A tile landed in LDS
#endif

    v8f acc[4] = {{}, {}, {}, {}};
    const unsigned int* B32 = (const unsigned int*)Xs;

    for (int cc = 0; cc < 4; ++cc) {             // ci chunks of 32
        __syncthreads();                         // A ready / Xs safe to overwrite
        int ci0 = cc * 32;
        // Stage x slab: rows hrow0-1..hrow0+2, cols -1..128, 32 channels (bf16).
        for (int i = tid; i < 4 * 130 * 32; i += 128) {
            int cr  = i / 130;
            int col = i - cr * 130;              // col fastest -> coalesced
            int c   = cr & 31;
            int r   = cr >> 5;
            int y   = hrow0 - 1 + r;
            int xx  = col - 1;
            float val = 0.f;
            if ((unsigned)y < H_ && (unsigned)xx < W_)
                val = x[((size_t)(b * CIN_ + ci0 + c) * H_ + y) * W_ + xx];
            Xs[(r * 130 + col) * 32 + c] = f2bf(val);
        }
        __syncthreads();

#pragma unroll
        for (int t = 0; t < 9; ++t) {            // taps: dy = t/3, dx = t%3
            int dy = t / 3, dx = t % 3;
            int kq = (t * 128 + cc * 32) >> 3;   // K/8 group base
            Frag av;                             // A 16-bit layout (ISA 7.12.2)
            av.q[0] = *(const uint4*)&AldsU[(kq + h) * 64 + m * 4];
            av.q[1] = *(const uint4*)&AldsU[(kq + 2 + h) * 64 + m * 4];
#pragma unroll
            for (int n = 0; n < 4; ++n) {        // B: lanes0-15 K=0..7,16..23, 16-31 K=8..15,24..31
                int base = ((prow[n] + dy) * 130 + pcol[n] + dx) * 16 + h * 4;
                Frag bv;
                bv.q[0] = *(const uint4*)&B32[base];
                bv.q[1] = *(const uint4*)&B32[base + 8];
                acc[n] = __builtin_amdgcn_wmma_f32_16x16x32_bf16(
                    false, av.b, false, bv.b, (short)0, acc[n], false, false);
            }
        }
    }

    // C layout: VGPR r -> co row r (lanes 0-15) / 8+r (lanes 16-31), N = lane&15.
#pragma unroll
    for (int n = 0; n < 4; ++n)
#pragma unroll
        for (int r = 0; r < 8; ++r) {
            int co = co0 + r + h * 8;
            float v = (acc[n][r] + bias[co]) * fbm[co];
            out[((size_t)(b * COUT_ + co) * H_ + hrow0 + prow[n]) * W_ + pcol[n]] = v;
        }
}

extern "C" void kernel_launch(void* const* d_in, const int* in_sizes, int n_in,
                              void* d_out, int out_size, void* d_ws, size_t ws_size,
                              hipStream_t stream) {
    (void)in_sizes; (void)n_in; (void)out_size; (void)ws_size;
    const float* x    = (const float*)d_in[0];
    const float* dftw = (const float*)d_in[1];
    const float* ksm  = (const float*)d_in[2];
    const float* bias = (const float*)d_in[3];
    const float* fbm  = (const float*)d_in[4];
    float* out = (float*)d_out;

    float* w      = (float*)d_ws;
    float* pooled = w;                               // B*CIN  = 2048 f
    float* att    = w + 2048;                        // B*KNUM = 1024 f
    unsigned short* kerb = (unsigned short*)(w + 4096); // B*8 tiles * 18432 bf16 = 4.7 MB

    fdconv_pool<<<dim3(B_ * CIN_), dim3(256), 0, stream>>>(x, pooled);
    fdconv_att<<<dim3(B_), dim3(64), 0, stream>>>(pooled, ksm, att);
    fdconv_agg_ifft<<<dim3((B_ * COUT_ * CIN_) / 256), dim3(256), 0, stream>>>(att, dftw, kerb);
    fdconv_conv_wmma<<<dim3(H_ / 2, COUT_ / 16, B_), dim3(128), 0, stream>>>(
        x, kerb, bias, fbm, out);
}
